// MultiHeadAttentionIFA_71476845740050
// MI455X (gfx1250) — matrix operand order repaired and checked by reading, not compiled, not run on hardware
//
#include <hip/hip_runtime.h>
#include <hip/hip_bf16.h>

// ---------------------------------------------------------------------------
// Types / helpers
// ---------------------------------------------------------------------------
typedef __attribute__((ext_vector_type(16))) __bf16 v16bf;
typedef __attribute__((ext_vector_type(8)))  float  v8f;
typedef __attribute__((ext_vector_type(4)))  unsigned int u32x4;

#define D_MODEL 1024
#define N_HEADS 16
#define D_HEAD  64
#define BATCH   2
#define SEQ     2048
#define M_TOT   (BATCH * SEQ)

// --- TDM availability -------------------------------------------------------
#if defined(__has_builtin)
#  if __has_builtin(__builtin_amdgcn_tensor_load_to_lds) && \
      __has_builtin(__builtin_amdgcn_s_wait_tensorcnt)
#    define USE_TDM 1
#  endif
#endif
#ifndef USE_TDM
#  define USE_TDM 0
#endif

#if USE_TDM
typedef __attribute__((ext_vector_type(4))) unsigned int tdm_g0_t;
typedef __attribute__((ext_vector_type(8))) int          tdm_g1_t;
typedef __attribute__((ext_vector_type(4))) int          tdm_g2_t;

// Issue a TDM 2D tile load: global (row-major, row stride in elements) -> LDS
// (packed row-major). Element size = 2 bytes (bf16). Descriptor per CDNA5 D#
// spec: group0 {count=1,lds_addr,global_addr,type=2}, group1 {data_size=2B,
// tensor dims, tile dims, dim0 stride}. Groups 2/3 zero (2D tensor); this
// toolchain's builtin takes 6 operands (extra int32x8 group), passed as zero.
__device__ __forceinline__ void tdm_load_2d_bf16(unsigned lds_off, const void* gptr,
                                                 unsigned tile_k, unsigned tile_rows,
                                                 unsigned tensor_d0, unsigned tensor_d1,
                                                 unsigned row_stride_elems) {
    unsigned long long ga = (unsigned long long)gptr;
    tdm_g0_t g0;
    g0[0] = 1u;                                   // count=1 (valid), flags 0
    g0[1] = lds_off;                              // lds_addr (bytes)
    g0[2] = (unsigned)ga;                         // global_addr[31:0]
    g0[3] = (unsigned)((ga >> 32) & 0x1FFFFFFu) | (2u << 30);  // addr[56:32] | type=2
    tdm_g1_t g1;
    g1[0] = (int)(1u << 16);                                       // data_size=2B
    g1[1] = (int)((tensor_d0 & 0xFFFFu) << 16);                    // tensor_dim0 lo
    g1[2] = (int)(((tensor_d0 >> 16) & 0xFFFFu) |
                  ((tensor_d1 & 0xFFFFu) << 16));                  // dim0 hi | dim1 lo
    g1[3] = (int)(((tensor_d1 >> 16) & 0xFFFFu) |
                  ((tile_k & 0xFFFFu) << 16));                     // dim1 hi | tile_dim0
    g1[4] = (int)(tile_rows & 0xFFFFu);                            // tile_dim1
    g1[5] = (int)row_stride_elems;                                 // tensor_dim0_stride
    g1[6] = 0;
    g1[7] = 0;
    tdm_g2_t z4 = (tdm_g2_t){0, 0, 0, 0};
    tdm_g1_t z8 = (tdm_g1_t){0, 0, 0, 0, 0, 0, 0, 0};
    __builtin_amdgcn_tensor_load_to_lds(g0, g1, z4, z4, z8, 0);
}
#endif

union Frag16 {
    v16bf v;
    unsigned short h[16];
    unsigned int   u[8];
};

__device__ __forceinline__ unsigned short f2bf(float x) {
    unsigned u = __float_as_uint(x);
    unsigned r = (u + 0x7FFFu + ((u >> 16) & 1u)) >> 16;   // round-nearest-even
    return (unsigned short)r;
}

__device__ __forceinline__ v8f wmma_bf16(const Frag16& a, const Frag16& b, v8f c) {
    return __builtin_amdgcn_wmma_f32_16x16x32_bf16(false, a.v, false, b.v,
                                                   (short)0, c, false, false);
}

// ---------------------------------------------------------------------------
// fp32 -> bf16 pack (vectorized by 4)
// ---------------------------------------------------------------------------
__global__ __launch_bounds__(256) void pack_bf16(const float* __restrict__ src,
                                                 unsigned short* __restrict__ dst,
                                                 int n4) {
    int i = blockIdx.x * 256 + threadIdx.x;
    if (i < n4) {
        float4 f = ((const float4*)src)[i];
        unsigned lo = (unsigned)f2bf(f.x) | ((unsigned)f2bf(f.y) << 16);
        unsigned hi = (unsigned)f2bf(f.z) | ((unsigned)f2bf(f.w) << 16);
        ((uint2*)dst)[i] = make_uint2(lo, hi);
    }
}

// ---------------------------------------------------------------------------
// GEMM: D(M x 1024) = A(M x 1024 bf16) @ W(1024 x 1024 bf16, row=out-col)^T + bias
// 128 threads = 4 waves; block tile 64(M) x 64(N); K chunks of 128, TDM
// double-buffered into 64KB dynamic LDS (wave 0 drives the DMA, TENSORcnt
// pipelined so chunk c+1 streams in while chunk c is consumed).
// ---------------------------------------------------------------------------
#define GK_CHUNK 128
#define GA_BUF   16384                      // bytes per A buffer (64*128*2)
#define GW_BASE  32768

template <bool OUTF32>
__global__ __launch_bounds__(128) void gemm_xwT(const unsigned short* __restrict__ A,
                                                const unsigned short* __restrict__ W,
                                                const float* __restrict__ bias,
                                                float* __restrict__ outF,
                                                unsigned short* __restrict__ outB) {
    extern __shared__ unsigned char smem[];
    const int tid  = threadIdx.x;
    const int wave = tid >> 5;
    const int lane = tid & 31;
    const int ln   = lane & 15;
    const int hi   = lane >> 4;
    const int tileM = blockIdx.x * 64;
    const int tileN = blockIdx.y * 64;
    const int NCH   = D_MODEL / GK_CHUNK;   // 8 chunks

    v8f c[4];
#pragma unroll
    for (int j = 0; j < 4; ++j) c[j] = (v8f){};

#if USE_TDM
    if (wave == 0) {   // prologue: chunk 0 -> buffer 0
        tdm_load_2d_bf16(0,       A + (size_t)tileM * D_MODEL, GK_CHUNK, 64,
                         D_MODEL, M_TOT, D_MODEL);
        tdm_load_2d_bf16(GW_BASE, W + (size_t)tileN * D_MODEL, GK_CHUNK, 64,
                         D_MODEL, D_MODEL, D_MODEL);
    }
#endif
    int buf = 0;
    for (int ch = 0; ch < NCH; ++ch) {
        const int k0 = ch * GK_CHUNK;
#if USE_TDM
        if (wave == 0) {
            if (ch + 1 < NCH) {
                const int kn = k0 + GK_CHUNK;
                tdm_load_2d_bf16((buf ^ 1) * GA_BUF,
                                 A + (size_t)tileM * D_MODEL + kn, GK_CHUNK, 64,
                                 D_MODEL, M_TOT, D_MODEL);
                tdm_load_2d_bf16(GW_BASE + (buf ^ 1) * GA_BUF,
                                 W + (size_t)tileN * D_MODEL + kn, GK_CHUNK, 64,
                                 D_MODEL, D_MODEL, D_MODEL);
                __builtin_amdgcn_s_wait_tensorcnt(2);   // chunk ch landed
            } else {
                __builtin_amdgcn_s_wait_tensorcnt(0);
            }
        }
#else
        {   // fallback: cooperative vector loads into buffer `buf`
            unsigned short* As = (unsigned short*)(smem + buf * GA_BUF);
            unsigned short* Ws = (unsigned short*)(smem + GW_BASE + buf * GA_BUF);
#pragma unroll
            for (int i = 0; i < 8; ++i) {
                int idx = tid + i * 128;            // 1024 uint4 per 64x128 tile
                int row = idx >> 4;
                int col = (idx & 15) * 8;
                *(u32x4*)(&As[row * GK_CHUNK + col]) =
                    *(const u32x4*)(&A[(size_t)(tileM + row) * D_MODEL + k0 + col]);
                *(u32x4*)(&Ws[row * GK_CHUNK + col]) =
                    *(const u32x4*)(&W[(size_t)(tileN + row) * D_MODEL + k0 + col]);
            }
        }
#endif
        __syncthreads();
        const unsigned short* As = (const unsigned short*)(smem + buf * GA_BUF);
        const unsigned short* Ws = (const unsigned short*)(smem + GW_BASE + buf * GA_BUF);
#pragma unroll
        for (int inner = 0; inner < GK_CHUNK / 32; ++inner) {
            const int koff = inner * 32;
            Frag16 a;
#pragma unroll
            for (int v = 0; v < 8; ++v) {
                int k = koff + (v >> 2) * 16 + hi * 8 + (v & 3) * 2;
                a.u[v] = *(const unsigned*)(&As[(wave * 16 + ln) * GK_CHUNK + k]);
            }
#pragma unroll
            for (int j = 0; j < 4; ++j) {
                Frag16 b;
#pragma unroll
                for (int v = 0; v < 8; ++v) {
                    int k = koff + (v >> 2) * 16 + hi * 8 + (v & 3) * 2;
                    b.u[v] = *(const unsigned*)(&Ws[(j * 16 + ln) * GK_CHUNK + k]);
                }
                c[j] = wmma_bf16(a, b, c[j]);
            }
        }
        __syncthreads();
        buf ^= 1;
    }

#pragma unroll
    for (int j = 0; j < 4; ++j) {
#pragma unroll
        for (int v = 0; v < 8; ++v) {
            int row = tileM + wave * 16 + v + hi * 8;
            int col = tileN + j * 16 + ln;
            float val = c[j][v] + bias[col];
            if (OUTF32) outF[(size_t)row * D_MODEL + col] = val;
            else        outB[(size_t)row * D_MODEL + col] = f2bf(val);
        }
    }
}

// ---------------------------------------------------------------------------
// Gate: sigmoid(q @ Wqh^T + bqh), 32 scalars (b,h)
// ---------------------------------------------------------------------------
__global__ __launch_bounds__(32) void gate_kernel(const float* __restrict__ q,
                                                  const float* __restrict__ Wqh,
                                                  const float* __restrict__ bqh,
                                                  float* __restrict__ gate) {
    int t = threadIdx.x;
    int b = t >> 4, h = t & 15;
    float acc = bqh[h];
    for (int i = 0; i < 64; ++i) acc += q[b * 64 + i] * Wqh[h * 64 + i];
    gate[t] = 1.0f / (1.0f + __expf(-acc));
}

// ---------------------------------------------------------------------------
// Flash-style attention. One wave per (b, h, 16-query tile), 32 keys/iter.
// S tile staged through LDS (transpose to A-layout); each lane then owns one
// full query row -> row max / row sum computed in-lane (no shuffles). Rescale
// factors broadcast back to C-layout through a 16-float LDS array. V chunks
// are TDM double-buffered; QK^T + softmax overlap the next V tile's DMA.
// uncertainty_bias is a mathematical no-op (added only to -inf logits).
// ---------------------------------------------------------------------------
#define AVS_BUF 4096                        // bytes per V buffer (32*64*2)
#define APS_OFF 8192                        // Ps: 16x32 f32
#define AFS_OFF 10240                       // Fs: 16 f32
#define ATTN_LDS 10304

__global__ __launch_bounds__(32) void attn_kernel(const unsigned short* __restrict__ Qb,
                                                  const unsigned short* __restrict__ Kb,
                                                  const unsigned short* __restrict__ Vb,
                                                  const int* __restrict__ mask,
                                                  const float* __restrict__ gate,
                                                  unsigned short* __restrict__ Ob) {
    extern __shared__ unsigned char smem[];
    float* Ps = (float*)(smem + APS_OFF);
    float* Fs = (float*)(smem + AFS_OFF);

    const int lane = threadIdx.x;
    const int ln   = lane & 15;
    const int hi   = lane >> 4;
    const int tile = blockIdx.x & 127;
    const int bh   = blockIdx.x >> 7;
    const int b    = bh >> 4;
    const int h    = bh & 15;
    const int rowbase = b * SEQ + tile * 16;
    const int hc      = h * D_HEAD;

    // Q fragments (d 0..31, 32..63) straight from global
    Frag16 qf[2];
#pragma unroll
    for (int ds = 0; ds < 2; ++ds)
#pragma unroll
        for (int v = 0; v < 8; ++v) {
            int k = ds * 32 + (v >> 2) * 16 + hi * 8 + (v & 3) * 2;
            qf[ds].u[v] = *(const unsigned*)(&Qb[(size_t)(rowbase + ln) * D_MODEL + hc + k]);
        }

    float mrow = -1e30f, lrow = 0.0f;       // this lane's query row (row = ln)
    v8f acc[4];
#pragma unroll
    for (int j = 0; j < 4; ++j) acc[j] = (v8f){};

#if USE_TDM
    tdm_load_2d_bf16(0, Vb + (size_t)(b * SEQ) * D_MODEL + hc, D_HEAD, 32,
                     D_HEAD, SEQ, D_MODEL);          // V chunk 0 -> buffer 0
#endif
    int buf = 0;
    for (int kc = 0; kc < SEQ / 32; ++kc) {
        const int key0 = kc * 32;
#if USE_TDM
        if (kc + 1 < SEQ / 32)
            tdm_load_2d_bf16((buf ^ 1) * AVS_BUF,
                             Vb + (size_t)(b * SEQ + key0 + 32) * D_MODEL + hc,
                             D_HEAD, 32, D_HEAD, SEQ, D_MODEL);
#else
        {
            unsigned short* Vs = (unsigned short*)(smem + buf * AVS_BUF);
#pragma unroll
            for (int i = 0; i < 8; ++i) {
                int idx = lane + i * 32;
                int r = idx >> 3, ccol = (idx & 7) * 8;
                *(u32x4*)(&Vs[r * 64 + ccol]) =
                    *(const u32x4*)(&Vb[(size_t)(b * SEQ + key0 + r) * D_MODEL + hc + ccol]);
            }
        }
#endif
        if (kc + 1 < SEQ / 32)   // prefetch next chunk's K rows
            __builtin_prefetch(&Kb[(size_t)(b * SEQ + key0 + 32 + lane) * D_MODEL + hc], 0, 1);

        // ---- S = Q @ K^T (two 16x16 key tiles), K frags gathered from global
        v8f s[2];
#pragma unroll
        for (int t = 0; t < 2; ++t) {
            v8f z = (v8f){};
#pragma unroll
            for (int ds = 0; ds < 2; ++ds) {
                Frag16 kb;
#pragma unroll
                for (int v = 0; v < 8; ++v) {
                    int d = ds * 32 + (v >> 2) * 16 + hi * 8 + (v & 3) * 2;
                    kb.u[v] = *(const unsigned*)(
                        &Kb[(size_t)(b * SEQ + key0 + t * 16 + ln) * D_MODEL + hc + d]);
                }
                z = wmma_bf16(qf[ds], kb, z);
            }
            s[t] = z;
        }

        // ---- scale + mask (lane = key column), stage into Ps transposed
        const int mk0 = mask[b * SEQ + key0 + ln];
        const int mk1 = mask[b * SEQ + key0 + 16 + ln];
#pragma unroll
        for (int v = 0; v < 8; ++v) {
            int r = v + hi * 8;
            Ps[r * 32 + ln]      = mk0 ? s[0][v] * 0.125f : -1e30f;
            Ps[r * 32 + 16 + ln] = mk1 ? s[1][v] * 0.125f : -1e30f;
        }
        __syncthreads();

        // ---- per-lane row softmax: lane owns row ln (both halves identical)
        float rowv[32];
        const float4* prow = (const float4*)(Ps + ln * 32);
#pragma unroll
        for (int i = 0; i < 8; ++i) {
            float4 t4 = prow[i];
            rowv[i * 4 + 0] = t4.x; rowv[i * 4 + 1] = t4.y;
            rowv[i * 4 + 2] = t4.z; rowv[i * 4 + 3] = t4.w;
        }
        float rmax = rowv[0];
#pragma unroll
        for (int i = 1; i < 32; ++i) rmax = fmaxf(rmax, rowv[i]);
        const float mnew = fmaxf(mrow, rmax);
        const float fscale = __expf(mrow - mnew);
        mrow = mnew;
        float rsum = 0.0f;
        float pexp[32];
#pragma unroll
        for (int i = 0; i < 32; ++i) { pexp[i] = __expf(rowv[i] - mnew); rsum += pexp[i]; }
        lrow = lrow * fscale + rsum;

        // P fragment is already in A-layout for this lane
        Frag16 pf;
#pragma unroll
        for (int v = 0; v < 8; ++v) {
            int k = (v >> 2) * 16 + hi * 8 + (v & 3) * 2;
            unsigned lo = f2bf(pexp[k]);
            unsigned up = f2bf(pexp[k + 1]);
            pf.u[v] = lo | (up << 16);
        }

        // broadcast per-row rescale factors to C-layout consumers
        if (hi == 0) Fs[ln] = fscale;
        __syncthreads();
        float Fv[8];
#pragma unroll
        for (int v = 0; v < 8; ++v) Fv[v] = Fs[v + 8 * hi];
#pragma unroll
        for (int j = 0; j < 4; ++j)
#pragma unroll
            for (int v = 0; v < 8; ++v) acc[j][v] *= Fv[v];

        // ---- P @ V: make sure current V buffer has landed, then 4 WMMAs
#if USE_TDM
        if (kc + 1 < SEQ / 32) __builtin_amdgcn_s_wait_tensorcnt(1);
        else                   __builtin_amdgcn_s_wait_tensorcnt(0);
#endif
        const unsigned short* Vs = (const unsigned short*)(smem + buf * AVS_BUF);
#pragma unroll
        for (int j = 0; j < 4; ++j) {
            Frag16 vb;
#pragma unroll
            for (int v = 0; v < 8; ++v) {
                int kk = (v >> 2) * 16 + hi * 8 + (v & 3) * 2;
                unsigned lo = Vs[kk * 64 + j * 16 + ln];
                unsigned up = Vs[(kk + 1) * 64 + j * 16 + ln];
                vb.u[v] = lo | (up << 16);
            }
            acc[j] = wmma_bf16(pf, vb, acc[j]);
        }
        __syncthreads();                   // protect Ps/Fs/Vs for next iter
        buf ^= 1;
    }

    // epilogue: broadcast row sums, normalize, gate, store bf16
    if (hi == 0) Fs[ln] = lrow;
    __syncthreads();
    const float g = gate[b * 16 + h];
#pragma unroll
    for (int j = 0; j < 4; ++j)
#pragma unroll
        for (int v = 0; v < 8; ++v) {
            float val = acc[j][v] / Fs[v + 8 * hi] * g;
            int row = rowbase + v + hi * 8;
            Ob[(size_t)row * D_MODEL + hc + j * 16 + ln] = f2bf(val);
        }
}

// ---------------------------------------------------------------------------
// Launch
// ---------------------------------------------------------------------------
extern "C" void kernel_launch(void* const* d_in, const int* in_sizes, int n_in,
                              void* d_out, int out_size, void* d_ws, size_t ws_size,
                              hipStream_t stream) {
    const float* x    = (const float*)d_in[0];
    const int*   mask = (const int*)  d_in[1];
    const float* q    = (const float*)d_in[2];
    const float* Wq   = (const float*)d_in[3];
    const float* bq   = (const float*)d_in[4];
    const float* Wk   = (const float*)d_in[5];
    const float* bk   = (const float*)d_in[6];
    const float* Wv   = (const float*)d_in[7];
    const float* bv   = (const float*)d_in[8];
    const float* Wo   = (const float*)d_in[9];
    const float* bo   = (const float*)d_in[10];
    const float* Wqh  = (const float*)d_in[12];
    const float* bqh  = (const float*)d_in[13];
    float* out = (float*)d_out;

    char* ws = (char*)d_ws;
    const size_t SZ_ACT = (size_t)M_TOT * D_MODEL * 2;
    const size_t SZ_W   = (size_t)D_MODEL * D_MODEL * 2;
    unsigned short* xb  = (unsigned short*)(ws);  ws += SZ_ACT;
    unsigned short* wqb = (unsigned short*)(ws);  ws += SZ_W;
    unsigned short* wkb = (unsigned short*)(ws);  ws += SZ_W;
    unsigned short* wvb = (unsigned short*)(ws);  ws += SZ_W;
    unsigned short* wob = (unsigned short*)(ws);  ws += SZ_W;
    unsigned short* Qb  = (unsigned short*)(ws);  ws += SZ_ACT;
    unsigned short* Kb  = (unsigned short*)(ws);  ws += SZ_ACT;
    unsigned short* Vb  = (unsigned short*)(ws);  ws += SZ_ACT;
    unsigned short* Ob  = (unsigned short*)(ws);  ws += SZ_ACT;
    float*          gate = (float*)(ws);          ws += 32 * sizeof(float);

    {
        int n4 = (M_TOT * D_MODEL) / 4;
        pack_bf16<<<(n4 + 255) / 256, 256, 0, stream>>>(x, xb, n4);
        int w4 = (D_MODEL * D_MODEL) / 4;
        int wg = (w4 + 255) / 256;
        pack_bf16<<<wg, 256, 0, stream>>>(Wq, wqb, w4);
        pack_bf16<<<wg, 256, 0, stream>>>(Wk, wkb, w4);
        pack_bf16<<<wg, 256, 0, stream>>>(Wv, wvb, w4);
        pack_bf16<<<wg, 256, 0, stream>>>(Wo, wob, w4);
    }

    dim3 ggrid(M_TOT / 64, D_MODEL / 64);
    const size_t GEMM_LDS = 65536;   // 2 x (64x128 A) + 2 x (64x128 W), bf16
    gemm_xwT<false><<<ggrid, 128, GEMM_LDS, stream>>>(xb, wqb, bq, nullptr, Qb);
    gemm_xwT<false><<<ggrid, 128, GEMM_LDS, stream>>>(xb, wkb, bk, nullptr, Kb);
    gemm_xwT<false><<<ggrid, 128, GEMM_LDS, stream>>>(xb, wvb, bv, nullptr, Vb);

    gate_kernel<<<1, 32, 0, stream>>>(q, Wqh, bqh, gate);

    attn_kernel<<<BATCH * N_HEADS * (SEQ / 16), 32, ATTN_LDS, stream>>>(
        Qb, Kb, Vb, mask, gate, Ob);

    gemm_xwT<true><<<ggrid, 128, GEMM_LDS, stream>>>(Ob, wob, bo, out, nullptr);
}
